// TFGPT2MainLayer_76510547411546
// MI455X (gfx1250) — matrix operand order repaired and checked by reading, not compiled, not run on hardware
//
#include <hip/hip_runtime.h>
#include <hip/hip_bf16.h>

// ---------------------------------------------------------------------------
// Types for CDNA5 WMMA
// ---------------------------------------------------------------------------
typedef __bf16 bf16;
typedef __attribute__((ext_vector_type(16))) __bf16 bf16x16;
typedef __attribute__((ext_vector_type(8)))  float   f32x8;

union Frag  { uint4 q[2]; bf16x16 v; };     // 32B fragment = 2 x b128
union Pack8 { uint4 q;    bf16 e[8]; };     // 8 bf16 = 1 x b128

__device__ __forceinline__ f32x8 wmma_bf16(bf16x16 a, bf16x16 b, f32x8 c) {
    return __builtin_amdgcn_wmma_f32_16x16x32_bf16(
        false, a, false, b, (short)0, c, false, false);
}

// ---------------------------------------------------------------------------
// Problem constants
// ---------------------------------------------------------------------------
#define BB   4
#define SS   1024
#define DD   768
#define HH   12
#define DHH  64
#define LL   6
#define TD   2304      // 3*D
#define DF   3072      // 4*D
#define MTOT 4096      // B*S

__device__ __forceinline__ float gelu_tanh(float x) {
    float x3 = x * x * x;
    return 0.5f * x * (1.0f + tanhf(0.7978845608028654f * (x + 0.044715f * x3)));
}

// ---------------------------------------------------------------------------
// 1) Embedding (float4 vectorized)
// ---------------------------------------------------------------------------
__global__ __launch_bounds__(256)
void embed_k(const int* __restrict__ ids, const int* __restrict__ rf,
             const float* __restrict__ wte, const float* __restrict__ wte_rf,
             const float* __restrict__ wpe, float* __restrict__ h)
{
    int idx = blockIdx.x * 256 + threadIdx.x;      // over B*S*D/4
    int d4  = idx % (DD / 4);
    int bs  = idx / (DD / 4);
    int s   = bs % SS;
    const float4 a = *(const float4*)&wte   [(size_t)ids[bs] * DD + d4 * 4];
    const float4 b = *(const float4*)&wte_rf[(size_t)rf [bs] * DD + d4 * 4];
    const float4 c = *(const float4*)&wpe   [(size_t)s       * DD + d4 * 4];
    float4 o; o.x = a.x+b.x+c.x; o.y = a.y+b.y+c.y; o.z = a.z+b.z+c.z; o.w = a.w+b.w+c.w;
    *(float4*)&h[(size_t)idx * 4] = o;
}

// ---------------------------------------------------------------------------
// 2) fp32 -> bf16 weight conversion, 8 elements/thread
// ---------------------------------------------------------------------------
__global__ __launch_bounds__(256)
void cvt_k(const float* __restrict__ x, bf16* __restrict__ y, size_t n8)
{
    size_t i = (size_t)blockIdx.x * 256 + threadIdx.x;
    if (i >= n8) return;
    const float4 f0 = *(const float4*)(x + i * 8);
    const float4 f1 = *(const float4*)(x + i * 8 + 4);
    Pack8 u;
    u.e[0] = (bf16)f0.x; u.e[1] = (bf16)f0.y; u.e[2] = (bf16)f0.z; u.e[3] = (bf16)f0.w;
    u.e[4] = (bf16)f1.x; u.e[5] = (bf16)f1.y; u.e[6] = (bf16)f1.z; u.e[7] = (bf16)f1.w;
    *(uint4*)(y + i * 8) = u.q;
}

// ---------------------------------------------------------------------------
// 3) LayerNorm: one 256-thread block per row (vectorized reads)
// ---------------------------------------------------------------------------
template <typename OUT>
__global__ __launch_bounds__(256)
void layernorm_k(const float* __restrict__ X, const float* __restrict__ g,
                 const float* __restrict__ b, OUT* __restrict__ Y)
{
    __shared__ float red[256];
    int row = blockIdx.x;
    int tid = threadIdx.x;
    const float* x = X + (size_t)row * DD;

    float s = 0.f, s2 = 0.f;
    for (int i = tid; i < DD / 4; i += 256) {
        float4 v = *(const float4*)&x[i * 4];
        s  += v.x + v.y + v.z + v.w;
        s2 += v.x*v.x + v.y*v.y + v.z*v.z + v.w*v.w;
    }
    red[tid] = s; __syncthreads();
    for (int off = 128; off > 0; off >>= 1) {
        if (tid < off) red[tid] += red[tid + off];
        __syncthreads();
    }
    float mu = red[0] * (1.0f / DD);
    __syncthreads();
    red[tid] = s2; __syncthreads();
    for (int off = 128; off > 0; off >>= 1) {
        if (tid < off) red[tid] += red[tid + off];
        __syncthreads();
    }
    float var  = red[0] * (1.0f / DD) - mu * mu;
    float rstd = rsqrtf(var + 1e-5f);

    OUT* y = Y + (size_t)row * DD;
    for (int i = tid; i < DD / 4; i += 256) {
        float4 v = *(const float4*)&x[i * 4];
        float4 gg = *(const float4*)&g[i * 4];
        float4 bb = *(const float4*)&b[i * 4];
        y[i*4+0] = (OUT)((v.x - mu) * rstd * gg.x + bb.x);
        y[i*4+1] = (OUT)((v.y - mu) * rstd * gg.y + bb.y);
        y[i*4+2] = (OUT)((v.z - mu) * rstd * gg.z + bb.z);
        y[i*4+3] = (OUT)((v.w - mu) * rstd * gg.w + bb.w);
    }
}

// ---------------------------------------------------------------------------
// 4) GEMM  C[M,N] = A[M,K](bf16) * W[K,N](bf16) + bias
//    Block tile 128x128x32; 8 waves in 4x2; wave tile 32x64 = 8 WMMAs/k-step
//    (2 A-frags x 4 B-frags, each B-frag reused twice). All traffic b128.
//    EPI 0: bf16 out | EPI 1: fp32 residual += | EPI 2: gelu -> bf16
// ---------------------------------------------------------------------------
template <int EPI>
__global__ __launch_bounds__(256)
void gemm_bf16(const bf16* __restrict__ A, const bf16* __restrict__ W,
               const float* __restrict__ bias, float* __restrict__ Hres,
               bf16* __restrict__ Obf, int N, int K)
{
    constexpr int LDA = 40;    // 80B rows
    constexpr int LDB = 136;   // 272B rows (128 + 8 pad), 16B aligned
    __shared__ __align__(16) bf16 As[128 * LDA];
    __shared__ __align__(16) bf16 Bs[32 * LDB];

    int tid   = threadIdx.x;
    int wave  = tid >> 5;
    int lane  = tid & 31;
    int mlane = lane & 15;
    int half  = lane >> 4;
    int wm    = wave & 3;            // row group (32 rows each)
    int wn    = wave >> 2;           // col group (64 cols each)
    int m0    = blockIdx.y * 128;
    int n0    = blockIdx.x * 128;

    f32x8 acc[2][4] = { { {}, {}, {}, {} }, { {}, {}, {}, {} } };

    for (int k0 = 0; k0 < K; k0 += 32) {
        __syncthreads();
        // Stage A: 128x32 halfs = 512 uint4, 2 per thread
        #pragma unroll
        for (int it = 0; it < 2; ++it) {
            int i = tid + it * 256;
            int r = i >> 2, c4 = i & 3;
            *(uint4*)&As[r * LDA + c4 * 8] =
                *(const uint4*)&A[(size_t)(m0 + r) * K + k0 + c4 * 8];
        }
        // Stage B: 32x128 halfs = 512 uint4, 2 per thread
        #pragma unroll
        for (int it = 0; it < 2; ++it) {
            int i = tid + it * 256;
            int r = i >> 4, c4 = i & 15;
            *(uint4*)&Bs[r * LDB + c4 * 8] =
                *(const uint4*)&W[(size_t)(k0 + r) * N + n0 + c4 * 8];
        }
        if (k0 + 32 < K) {
            __builtin_prefetch(&A[(size_t)(m0 + (tid >> 2)) * K + k0 + 32 + (tid & 3) * 8], 0, 3);
            __builtin_prefetch(&W[(size_t)(k0 + 32 + (tid >> 4)) * N + n0 + (tid & 15) * 8], 0, 3);
        }
        __syncthreads();

        // Two A fragments (row groups wm*32 and wm*32+16)
        Frag a0, a1;
        {
            const bf16* ar0 = &As[(wm * 32 + mlane) * LDA];
            const bf16* ar1 = &As[(wm * 32 + 16 + mlane) * LDA];
            a0.q[0] = *(const uint4*)&ar0[half * 8];
            a0.q[1] = *(const uint4*)&ar0[16 + half * 8];
            a1.q[0] = *(const uint4*)&ar1[half * 8];
            a1.q[1] = *(const uint4*)&ar1[16 + half * 8];
        }
        // Four B fragments, each reused for both A row groups
        const bf16* brow = &Bs[wn * 64 + mlane];
        #pragma unroll
        for (int t = 0; t < 4; ++t) {
            Frag b;
            #pragma unroll
            for (int s = 0; s < 16; ++s)
                b.v[s] = brow[((s & 7) + half * 8 + (s >> 3) * 16) * LDB + t * 16];
            acc[0][t] = wmma_bf16(a0.v, b.v, acc[0][t]);
            acc[1][t] = wmma_bf16(a1.v, b.v, acc[1][t]);
        }
    }

    // C/D layout: vgpr r -> M = r + 8*half ; N = lane%16
    #pragma unroll
    for (int g = 0; g < 2; ++g)
        #pragma unroll
        for (int t = 0; t < 4; ++t)
            #pragma unroll
            for (int r = 0; r < 8; ++r) {
                int row = m0 + wm * 32 + g * 16 + r + 8 * half;
                int col = n0 + wn * 64 + t * 16 + mlane;
                float v = acc[g][t][r] + bias[col];
                size_t i0 = (size_t)row * N + col;
                if (EPI == 1) {
                    Hres[i0] += v;
                } else {
                    if (EPI == 2) v = gelu_tanh(v);
                    Obf[i0] = (bf16)v;
                }
            }
}

// ---------------------------------------------------------------------------
// 5) Fused causal flash attention, TRANSPOSED formulation.
//    Grid (S/64, H, B); 128 threads = 4 waves; wave w owns q cols [w*16,+16).
//    S' = K * Q^T  (C-layout: M=key, N=q)  -> softmax reduction is in-lane
//    (plus a single shfl_xor(16)); O^T = V^T * P' accumulated per dh-tile.
// ---------------------------------------------------------------------------
__global__ __launch_bounds__(128)
void attn_k(const bf16* __restrict__ qkv, bf16* __restrict__ out)
{
    constexpr int LDQ = 72;   // Qt[dh=64][q=64+pad]
    constexpr int LDKS = 72;  // Ks[key=32][dh=64+pad]
    constexpr int LDVT = 40;  // Vt[dh=64][key=32+pad]
    constexpr int LDP = 72;   // Ps[key=32][q=64+pad]
    __shared__ __align__(16) bf16 Qt[64 * LDQ];
    __shared__ __align__(16) bf16 Ks[32 * LDKS];
    __shared__ __align__(16) bf16 Vt[64 * LDVT];
    __shared__ __align__(16) bf16 Ps[32 * LDP];

    int tid   = threadIdx.x;
    int wave  = tid >> 5;
    int lane  = tid & 31;
    int mlane = lane & 15;
    int half  = lane >> 4;

    int qb    = blockIdx.x;
    int head  = blockIdx.y;
    int b     = blockIdx.z;
    int qrow0 = qb * 64;
    int qw0   = wave * 16;              // wave's q columns within block
    int hoff  = head * DHH;
    size_t base = (size_t)b * SS * TD;

    // Stage Q^T once: Qt[dh][q] from 64 q-rows x 64 dh
    #pragma unroll
    for (int it = 0; it < 4; ++it) {
        int i = tid + it * 128;         // 0..511
        int r = i >> 3, c4 = i & 7;     // q row, dh chunk
        Pack8 u;
        u.q = *(const uint4*)&qkv[base + (size_t)(qrow0 + r) * TD + hoff + c4 * 8];
        #pragma unroll
        for (int j = 0; j < 8; ++j)
            Qt[(c4 * 8 + j) * LDQ + r] = u.e[j];
    }

    float mrun = -1e30f, lrun = 0.f;
    f32x8 o[4] = { {}, {}, {}, {} };
    int qg = qrow0 + qw0 + mlane;       // this lane's q index

    int nblk = (qb + 1) * 2;            // 32-key causal blocks
    for (int jb = 0; jb < nblk; ++jb) {
        int j0 = jb * 32;
        __syncthreads();
        // Stage K row-major (vector) + V transposed (scatter)
        #pragma unroll
        for (int it = 0; it < 2; ++it) {
            int i = tid + it * 128;
            int r = i >> 3, c4 = i & 7;              // key, dh chunk
            size_t srow = base + (size_t)(j0 + r) * TD;
            *(uint4*)&Ks[r * LDKS + c4 * 8] =
                *(const uint4*)&qkv[srow + DD + hoff + c4 * 8];
            Pack8 u;
            u.q = *(const uint4*)&qkv[srow + 2 * DD + hoff + c4 * 8];
            #pragma unroll
            for (int j = 0; j < 8; ++j)
                Vt[(c4 * 8 + j) * LDVT + r] = u.e[j];
        }
        __syncthreads();

        // Q^T B-frags (shared across both key tiles): lane = q column, elements = dh
        Frag qb0, qb1;
        #pragma unroll
        for (int s = 0; s < 16; ++s) {
            int kk = (s & 7) + half * 8 + (s >> 3) * 16;
            qb0.v[s] = Qt[kk * LDQ + qw0 + mlane];
            qb1.v[s] = Qt[(32 + kk) * LDQ + qw0 + mlane];
        }

        // S'[key][q] per 16-key tile: A = K rows (row-major, contiguous)
        f32x8 sc[2];
        #pragma unroll
        for (int t = 0; t < 2; ++t) {
            const bf16* krow = &Ks[(t * 16 + mlane) * LDKS];
            Frag a0, a1;
            a0.q[0] = *(const uint4*)&krow[half * 8];
            a0.q[1] = *(const uint4*)&krow[16 + half * 8];
            a1.q[0] = *(const uint4*)&krow[32 + half * 8];
            a1.q[1] = *(const uint4*)&krow[48 + half * 8];
            f32x8 z = {};
            z = wmma_bf16(a0.v, qb0.v, z);
            z = wmma_bf16(a1.v, qb1.v, z);
            sc[t] = z;
        }

        // scale + causal mask (element: key = j0+t*16+half*8+r, q = qg)
        bool need_mask = (j0 + 31) > (qrow0 + qw0);   // wave-uniform
        #pragma unroll
        for (int t = 0; t < 2; ++t)
            #pragma unroll
            for (int r = 0; r < 8; ++r) {
                float v  = sc[t][r] * 0.125f;
                if (need_mask) {
                    int kg = j0 + t * 16 + half * 8 + r;
                    v = (kg <= qg) ? v : -10000.0f;
                }
                sc[t][r] = v;
            }

        // online softmax: in-lane tree + ONE cross-half shuffle each
        float bm = sc[0][0];
        #pragma unroll
        for (int r = 1; r < 8; ++r) bm = fmaxf(bm, sc[0][r]);
        #pragma unroll
        for (int r = 0; r < 8; ++r) bm = fmaxf(bm, sc[1][r]);
        bm = fmaxf(bm, __shfl_xor(bm, 16, 32));
        float mnew  = fmaxf(mrun, bm);
        float alpha = __expf(mrun - mnew);

        float p[2][8];
        float ps = 0.f;
        #pragma unroll
        for (int t = 0; t < 2; ++t)
            #pragma unroll
            for (int r = 0; r < 8; ++r) {
                float e = __expf(sc[t][r] - mnew);
                p[t][r] = e;
                ps += e;
            }
        ps += __shfl_xor(ps, 16, 32);
        lrun = lrun * alpha + ps;
        mrun = mnew;
        #pragma unroll
        for (int t = 0; t < 4; ++t) o[t] *= alpha;

        // store P' (already in [key][q] layout, per-wave disjoint q columns)
        #pragma unroll
        for (int t = 0; t < 2; ++t)
            #pragma unroll
            for (int r = 0; r < 8; ++r)
                Ps[(t * 16 + half * 8 + r) * LDP + qw0 + mlane] = (bf16)p[t][r];
        __syncthreads();

        // O^T[dh][q] += V^T(16dh x 32key) * P'(32key x 16q)
        Frag pb;
        #pragma unroll
        for (int s = 0; s < 16; ++s)
            pb.v[s] = Ps[((s & 7) + half * 8 + (s >> 3) * 16) * LDP + qw0 + mlane];
        #pragma unroll
        for (int t = 0; t < 4; ++t) {
            const bf16* vrow = &Vt[(t * 16 + mlane) * LDVT];
            Frag va;
            va.q[0] = *(const uint4*)&vrow[half * 8];
            va.q[1] = *(const uint4*)&vrow[16 + half * 8];
            o[t] = wmma_bf16(va.v, pb.v, o[t]);
        }
    }

    // normalize (per-lane 1/l) and scatter O^T -> out[B,S,D]
    float rl = 1.0f / lrun;
    #pragma unroll
    for (int t = 0; t < 4; ++t)
        #pragma unroll
        for (int r = 0; r < 8; ++r) {
            int dh = t * 16 + r + 8 * half;
            out[((size_t)(b * SS + qg)) * DD + hoff + dh] = (bf16)(o[t][r] * rl);
        }
}

// ---------------------------------------------------------------------------
// Host launcher
// ---------------------------------------------------------------------------
extern "C" void kernel_launch(void* const* d_in, const int* in_sizes, int n_in,
                              void* d_out, int out_size, void* d_ws, size_t ws_size,
                              hipStream_t stream)
{
    (void)in_sizes; (void)n_in; (void)out_size; (void)ws_size;

    const int*   input_ids = (const int*)d_in[0];
    const int*   rf_ids    = (const int*)d_in[1];
    const float* wte       = (const float*)d_in[2];
    const float* wte_rf    = (const float*)d_in[3];
    const float* wpe       = (const float*)d_in[4];
    const float* ln1_g     = (const float*)d_in[5];
    const float* ln1_b     = (const float*)d_in[6];
    const float* attn_w    = (const float*)d_in[7];
    const float* attn_b    = (const float*)d_in[8];
    const float* proj_w    = (const float*)d_in[9];
    const float* proj_b    = (const float*)d_in[10];
    const float* ln2_g     = (const float*)d_in[11];
    const float* ln2_b     = (const float*)d_in[12];
    const float* fc_w      = (const float*)d_in[13];
    const float* fc_b      = (const float*)d_in[14];
    const float* fc2_w     = (const float*)d_in[15];
    const float* fc2_b     = (const float*)d_in[16];
    const float* lnf_g     = (const float*)d_in[17];
    const float* lnf_b     = (const float*)d_in[18];

    char* p = (char*)d_ws;
    auto carve = [&](size_t bytes) -> void* {
        void* r = (void*)p;
        p += (bytes + 255) & ~(size_t)255;
        return r;
    };
    float* h    = (float*)carve((size_t)MTOT * DD * 4);
    bf16*  xb   = (bf16*) carve((size_t)MTOT * DD * 2);
    bf16*  qkv  = (bf16*) carve((size_t)MTOT * TD * 2);
    bf16*  att  = (bf16*) carve((size_t)MTOT * DD * 2);
    bf16*  mlp  = (bf16*) carve((size_t)MTOT * DF * 2);
    bf16*  wqkv = (bf16*) carve((size_t)LL * DD * TD * 2);
    bf16*  wprj = (bf16*) carve((size_t)LL * DD * DD * 2);
    bf16*  wfc  = (bf16*) carve((size_t)LL * DD * DF * 2);
    bf16*  wfc2 = (bf16*) carve((size_t)LL * DF * DD * 2);

    embed_k<<<(MTOT * DD / 4) / 256, 256, 0, stream>>>(input_ids, rf_ids, wte, wte_rf, wpe, h);

    auto cvt = [&](const float* src, bf16* dst, size_t n) {
        size_t n8 = n / 8;
        cvt_k<<<(unsigned)((n8 + 255) / 256), 256, 0, stream>>>(src, dst, n8);
    };
    cvt(attn_w, wqkv, (size_t)LL * DD * TD);
    cvt(proj_w, wprj, (size_t)LL * DD * DD);
    cvt(fc_w,   wfc,  (size_t)LL * DD * DF);
    cvt(fc2_w,  wfc2, (size_t)LL * DF * DD);

    dim3 gQKV(TD / 128, MTOT / 128);
    dim3 gPRJ(DD / 128, MTOT / 128);
    dim3 gFC (DF / 128, MTOT / 128);
    dim3 gATT(SS / 64, HH, BB);

    for (int l = 0; l < LL; ++l) {
        layernorm_k<bf16><<<MTOT, 256, 0, stream>>>(h, ln1_g + l * DD, ln1_b + l * DD, xb);
        gemm_bf16<0><<<gQKV, 256, 0, stream>>>(xb, wqkv + (size_t)l * DD * TD,
                                               attn_b + (size_t)l * TD, nullptr, qkv, TD, DD);
        attn_k<<<gATT, 128, 0, stream>>>(qkv, att);
        gemm_bf16<1><<<gPRJ, 256, 0, stream>>>(att, wprj + (size_t)l * DD * DD,
                                               proj_b + (size_t)l * DD, h, nullptr, DD, DD);
        layernorm_k<bf16><<<MTOT, 256, 0, stream>>>(h, ln2_g + l * DD, ln2_b + l * DD, xb);
        gemm_bf16<2><<<gFC, 256, 0, stream>>>(xb, wfc + (size_t)l * DD * DF,
                                              fc_b + (size_t)l * DF, nullptr, mlp, DF, DD);
        gemm_bf16<1><<<gPRJ, 256, 0, stream>>>(mlp, wfc2 + (size_t)l * DF * DD,
                                               fc2_b + (size_t)l * DD, h, nullptr, DD, DF);
    }

    layernorm_k<float><<<MTOT, 256, 0, stream>>>(h, lnf_g, lnf_b, (float*)d_out);
}
